// StellarByteAttention_23742579212899
// MI455X (gfx1250) — matrix operand order repaired and checked by reading, not compiled, not run on hardware
//
#include <hip/hip_runtime.h>

// ---------------------------------------------------------------------------
// StellarByteAttention for MI455X (gfx1250, wave32, WMMA bf16->f32)
// Pipeline: f32->bf16 convert/transpose -> WMMA GEMMs (QKV) with async-to-LDS
//           double buffering -> RoPE -> flash attention (async-LDS K/V tiles,
//           WMMA QK^T + PV, online softmax) -> WMMA out-proj
// ---------------------------------------------------------------------------

typedef __attribute__((ext_vector_type(16))) __bf16 bf16x16;
typedef __attribute__((ext_vector_type(8)))  __bf16 bf16x8;
typedef __attribute__((ext_vector_type(8)))  float  f32x8;

#define S_LEN   2048
#define NHEADS  32
#define KVHEADS 8
#define HDIM    128
#define DMODEL  4096
#define BATCH   2
#define BSROWS  (BATCH * S_LEN)     // 4096
#define QCOLS   (NHEADS * HDIM)     // 4096
#define KVCOLS  (KVHEADS * HDIM)    // 1024

#define LDS_STRIDE 40               // 32 K elems padded to 40 (80B) per row
#define KT_STRIDE  136              // 128 d elems padded to 136 (272B) per key

// ---------------- bf16 helpers (round-to-nearest-even) ----------------------
__device__ __forceinline__ __bf16 f2bf(float f) {
  unsigned u = __builtin_bit_cast(unsigned, f);
  unsigned r = u + 0x7FFFu + ((u >> 16) & 1u);
  unsigned short h = (unsigned short)(r >> 16);
  return __builtin_bit_cast(__bf16, h);
}
__device__ __forceinline__ float bf2f(__bf16 b) {
  unsigned short h = __builtin_bit_cast(unsigned short, b);
  unsigned u = ((unsigned)h) << 16;
  return __builtin_bit_cast(float, u);
}

// ---------------- WMMA fragment loads (ISA 7.12.2 layouts) ------------------
// A 16x32 bf16, row-major, stride lda (elements):
//   lane<16 : elems 0..7 = K(0..7),  elems 8..15 = K(16..23)
//   lane>=16: elems 0..7 = K(8..15), elems 8..15 = K(24..31)
__device__ __forceinline__ bf16x16 load_a_frag(const __bf16* a, int lda) {
  int lane = threadIdx.x & 31;
  int row  = lane & 15;
  int koff = (lane >> 4) * 8;
  bf16x8 lo = *(const bf16x8*)(a + (size_t)row * lda + koff);
  bf16x8 hi = *(const bf16x8*)(a + (size_t)row * lda + 16 + koff);
  bf16x16 r;
#pragma unroll
  for (int i = 0; i < 8; ++i) { r[i] = lo[i]; r[8 + i] = hi[i]; }
  return r;
}

// B 32x16 bf16 supplied TRANSPOSED: bt[n][k], stride ldb; 32-byte aligned rows
__device__ __forceinline__ bf16x16 load_b_frag(const __bf16* bt, int ldb) {
  int lane = threadIdx.x & 31;
  int col  = lane & 15;
  int kb   = (lane >> 4) * 16;
  return *(const bf16x16*)(bt + (size_t)col * ldb + kb);
}

// Like load_b_frag but in load_a_frag's k order, via two 16B reads (for LDS rows with 16B-mult pad)
__device__ __forceinline__ bf16x16 load_b_frag16(const __bf16* bt, int ldb) {
  int lane = threadIdx.x & 31;
  int col  = lane & 15;
  int kb   = (lane >> 4) * 8;
  const __bf16* p = bt + (size_t)col * ldb + kb;
  bf16x8 lo = *(const bf16x8*)p;
  bf16x8 hi = *(const bf16x8*)(p + 16);
  bf16x16 r;
#pragma unroll
  for (int i = 0; i < 8; ++i) { r[i] = lo[i]; r[8 + i] = hi[i]; }
  return r;
}

__device__ __forceinline__ f32x8 wmma_bf16(bf16x16 a, bf16x16 b, f32x8 c) {
  return __builtin_amdgcn_wmma_f32_16x16x32_bf16(false, a, false, b,
                                                 (short)0, c, false, false);
}

// ---------------- CDNA5 async global->LDS copy (ASYNCcnt path) --------------
__device__ __forceinline__ void async_load_b128(const __bf16* lds_dst,
                                                const __bf16* gsrc) {
  unsigned loff = (unsigned)(unsigned long long)lds_dst;   // LDS byte offset
  unsigned long long ga = (unsigned long long)gsrc;
  asm volatile("global_load_async_to_lds_b128 %0, %1, off"
               :: "v"(loff), "v"(ga)
               : "memory");
}
__device__ __forceinline__ void wait_async0() {
  asm volatile("s_wait_asynccnt 0" ::: "memory");
}

// ---------------- prep kernels ----------------------------------------------
__global__ void k_f32_to_bf16(const float* __restrict__ src,
                              __bf16* __restrict__ dst, int n) {
  int i = blockIdx.x * blockDim.x + threadIdx.x;
  if (i < n) dst[i] = f2bf(src[i]);
}

// dst[c*R + r] = bf16(src[r*C + c])   (W [K][N] -> Wt [N][K])
__global__ void k_transpose_f32_bf16(const float* __restrict__ src,
                                     __bf16* __restrict__ dst, int R, int C) {
  int i = blockIdx.x * blockDim.x + threadIdx.x;
  if (i >= R * C) return;
  int r = i / C, c = i % C;
  dst[(size_t)c * R + r] = f2bf(src[i]);
}

// interleaved-pair RoPE on bf16 activations, layout [BSROWS][nh*HDIM]
__global__ void k_rope(__bf16* __restrict__ t, const float* __restrict__ cosb,
                       const float* __restrict__ sinb, int nh, int total) {
  int g = blockIdx.x * blockDim.x + threadIdx.x;
  if (g >= total) return;
  int i    = g & 63;           // pair index within head dim
  int rest = g >> 6;
  int head = rest % nh;
  int row  = rest / nh;        // b*S + s
  int s    = row & (S_LEN - 1);
  float c  = cosb[s * 64 + i];
  float sn = sinb[s * 64 + i];
  size_t base = (size_t)row * (nh * HDIM) + head * HDIM + 2 * i;
  float t1 = bf2f(t[base]), t2 = bf2f(t[base + 1]);
  t[base]     = f2bf(t1 * c - t2 * sn);
  t[base + 1] = f2bf(t1 * sn + t2 * c);
}

// ---------------- WMMA GEMM: C = A(bf16) x Bt(bf16)^T -----------------------
// 256 threads = 8 waves (4 M-slices x 2 N-slices).
// Block tile 128x128, wave tile 32x64, K-step 32.
// A/B K-tiles staged in LDS via global_load_async_to_lds_b128, double-buffered.
// mode 0: bf16 row-major store   mode 1: bf16 per-batch transposed (V^T)
// mode 2: f32 row-major store (final output)
__global__ __launch_bounds__(256) void k_gemm_bf16(
    const __bf16* __restrict__ A, const __bf16* __restrict__ Bt,
    void* __restrict__ Cout, int M, int N, int K, int mode) {
  // per buffer: A tile 128x40 + B tile 128x40 bf16 = 20480 B; x2 buffers
  __shared__ __align__(16) __bf16 smem[2][2 * 128 * LDS_STRIDE];

  const int tid  = threadIdx.x;
  const int lane = tid & 31;
  const int wave = tid >> 5;
  const int wm = wave & 3, wn = wave >> 2;
  const int rowblk = blockIdx.y * 128;
  const int colblk = blockIdx.x * 128;

  // cooperative copy mapping: 2 x b128 per 32-elem row, 128 rows per matrix
  const int r2   = tid >> 1;        // 0..127 : row of A tile / col of B tile
  const int half = tid & 1;         // which 16B chunk of the 64B row

  f32x8 acc[2][4] = {};

  const int nk = K >> 5;
  // prologue: issue K-tile 0 into buffer 0
  {
    const __bf16* ga = A  + (size_t)(rowblk + r2) * K + half * 8;
    const __bf16* gb = Bt + (size_t)(colblk + r2) * K + half * 8;
    async_load_b128(&smem[0][r2 * LDS_STRIDE + half * 8], ga);
    async_load_b128(&smem[0][128 * LDS_STRIDE + r2 * LDS_STRIDE + half * 8], gb);
  }

  for (int ik = 0; ik < nk; ++ik) {
    const int cur = ik & 1;
    wait_async0();        // our async writes into buf[cur] have landed
    __syncthreads();      // everyone's writes landed; buf[cur^1] readers done

    if (ik + 1 < nk) {    // prefetch next K-tile into the other buffer
      int kk = (ik + 1) << 5;
      const __bf16* ga = A  + (size_t)(rowblk + r2) * K + kk + half * 8;
      const __bf16* gb = Bt + (size_t)(colblk + r2) * K + kk + half * 8;
      async_load_b128(&smem[cur ^ 1][r2 * LDS_STRIDE + half * 8], ga);
      async_load_b128(&smem[cur ^ 1][128 * LDS_STRIDE + r2 * LDS_STRIDE + half * 8], gb);
    }

    const __bf16* ldsA = &smem[cur][0];
    const __bf16* ldsB = &smem[cur][128 * LDS_STRIDE];
    bf16x16 af0 = load_a_frag(ldsA + (wm * 32 + 0)  * LDS_STRIDE, LDS_STRIDE);
    bf16x16 af1 = load_a_frag(ldsA + (wm * 32 + 16) * LDS_STRIDE, LDS_STRIDE);
#pragma unroll
    for (int t = 0; t < 4; ++t) {
      bf16x16 bfr = load_b_frag16(ldsB + (wn * 64 + t * 16) * LDS_STRIDE, LDS_STRIDE);
      acc[0][t] = wmma_bf16(af0, bfr, acc[0][t]);
      acc[1][t] = wmma_bf16(af1, bfr, acc[1][t]);
    }
  }

  const int coln  = lane & 15;
  const int rbase = (lane >> 4) * 8;
#pragma unroll
  for (int mi = 0; mi < 2; ++mi) {
#pragma unroll
    for (int t = 0; t < 4; ++t) {
#pragma unroll
      for (int j = 0; j < 8; ++j) {
        int r = rowblk + wm * 32 + mi * 16 + rbase + j;
        int c = colblk + wn * 64 + t * 16 + coln;
        float v = acc[mi][t][j];
        if (mode == 2) {
          ((float*)Cout)[(size_t)r * N + c] = v;
        } else if (mode == 1) {            // V^T: [b][col][s], S_LEN=2048
          int b = r >> 11, s = r & (S_LEN - 1);
          ((__bf16*)Cout)[((size_t)b * N + c) * S_LEN + s] = f2bf(v);
        } else {
          ((__bf16*)Cout)[(size_t)r * N + c] = f2bf(v);
        }
      }
    }
  }
}

// ---------------- flash attention (causal, GQA 4:1) -------------------------
// 128 threads = 4 waves; each wave owns 16 query rows (block covers 64).
// K/V tiles (32 keys) are staged block-wide in LDS with async copies,
// double-buffered; waves run a uniform barrier loop and skip compute for
// fully-masked key blocks.
#define KT_ELEMS (32 * KT_STRIDE)        // K tile: 32 keys x 128 d (padded)
#define VT_ELEMS (128 * LDS_STRIDE)      // V^T tile: 128 d x 32 keys (padded)
__global__ __launch_bounds__(128) void k_flash_attn(
    const __bf16* __restrict__ Q, const __bf16* __restrict__ Kc,
    const __bf16* __restrict__ Vt, __bf16* __restrict__ ctx) {
  __shared__ __align__(16) __bf16 kv_smem[2][KT_ELEMS + VT_ELEMS];
  __shared__ unsigned short lds_raw[4 * 16 * 32];   // per-wave P staging (bf16)

  const int tid  = threadIdx.x;
  const int lane = tid & 31;
  const int wave = tid >> 5;
  __bf16* plds = (__bf16*)(lds_raw + wave * 512);

  const int b = blockIdx.z, h = blockIdx.y;
  const int kvh = h >> 2;
  const int qbase = (blockIdx.x * 4 + wave) * 16;
  const float scale = 0.08838834764831845f;   // 1/sqrt(128)

  const int coln  = lane & 15;
  const int rbase = (lane >> 4) * 8;

  // Q fragments for this wave's 16 rows x 128 dims (4 x K=32 chunks)
  const __bf16* qp = Q + ((size_t)(b * S_LEN + qbase)) * QCOLS + h * HDIM;
  bf16x16 qf[4];
#pragma unroll
  for (int c = 0; c < 4; ++c) qf[c] = load_a_frag(qp + c * 32, QCOLS);

  f32x8 o[8] = {};
  float mrow[8], lrow[8];
#pragma unroll
  for (int j = 0; j < 8; ++j) { mrow[j] = -1e30f; lrow[j] = 0.f; }

  const __bf16* kp    = Kc + ((size_t)b * S_LEN) * KVCOLS + kvh * HDIM;
  const __bf16* vbase = Vt + ((size_t)(b * KVHEADS + kvh)) * HDIM * S_LEN;

  const int hi_row  = qbase + 15;
  const int nblocks = (blockIdx.x + 1) * 2;   // keys 0 .. blockIdx.x*64+63

  // prologue: stage key block 0 into buffer 0 (8 async b128 per thread)
  {
#pragma unroll
    for (int i = 0; i < 4; ++i) {               // K tile: 32 x 256B
      int cid = tid + i * 128, krow = cid >> 4, kc = cid & 15;
      async_load_b128(&kv_smem[0][krow * KT_STRIDE + kc * 8],
                      kp + (size_t)krow * KVCOLS + kc * 8);
    }
#pragma unroll
    for (int i = 0; i < 4; ++i) {               // V^T tile: 128 x 64B
      int cid = tid + i * 128, d = cid >> 2, c = cid & 3;
      async_load_b128(&kv_smem[0][KT_ELEMS + d * LDS_STRIDE + c * 8],
                      vbase + (size_t)d * S_LEN + c * 8);
    }
  }

  for (int ib = 0; ib < nblocks; ++ib) {
    const int kb  = ib * 32;
    const int cur = ib & 1;
    wait_async0();        // our async writes into buf[cur] landed
    __syncthreads();      // all writes landed; buf[cur^1] readers finished

    if (ib + 1 < nblocks) {   // prefetch next key block into other buffer
      int kn = kb + 32;
#pragma unroll
      for (int i = 0; i < 4; ++i) {
        int cid = tid + i * 128, krow = cid >> 4, kc = cid & 15;
        async_load_b128(&kv_smem[cur ^ 1][krow * KT_STRIDE + kc * 8],
                        kp + (size_t)(kn + krow) * KVCOLS + kc * 8);
      }
#pragma unroll
      for (int i = 0; i < 4; ++i) {
        int cid = tid + i * 128, d = cid >> 2, c = cid & 3;
        async_load_b128(&kv_smem[cur ^ 1][KT_ELEMS + d * LDS_STRIDE + c * 8],
                        vbase + (size_t)d * S_LEN + kn + c * 8);
      }
    }

    if (kb > hi_row) continue;      // fully-masked block: barriers done above

    const __bf16* kbuf = &kv_smem[cur][0];
    const __bf16* vbuf = &kv_smem[cur][KT_ELEMS];

    // ---- scores: 16 q-rows x 32 keys, contraction over HDIM=128 ----
    f32x8 s0 = {}, s1 = {};
#pragma unroll
    for (int c = 0; c < 4; ++c) {
      bf16x16 k0 = load_b_frag16(kbuf + c * 32, KT_STRIDE);
      bf16x16 k1 = load_b_frag16(kbuf + 16 * KT_STRIDE + c * 32, KT_STRIDE);
      s0 = wmma_bf16(qf[c], k0, s0);
      s1 = wmma_bf16(qf[c], k1, s1);
    }

    // ---- scale + causal mask + row max ----
    float mnew[8];
#pragma unroll
    for (int j = 0; j < 8; ++j) {
      int r = qbase + rbase + j;
      float v0 = s0[j] * scale;
      float v1 = s1[j] * scale;
      if (kb + coln > r)      v0 = -1e30f;
      if (kb + 16 + coln > r) v1 = -1e30f;
      s0[j] = v0; s1[j] = v1;
      mnew[j] = fmaxf(v0, v1);
    }
#pragma unroll
    for (int off = 8; off >= 1; off >>= 1)
#pragma unroll
      for (int j = 0; j < 8; ++j)
        mnew[j] = fmaxf(mnew[j], __shfl_xor(mnew[j], off, 32));

    // ---- online softmax update ----
    float alpha[8], rs[8];
#pragma unroll
    for (int j = 0; j < 8; ++j) {
      float mi = fmaxf(mrow[j], mnew[j]);
      alpha[j] = __expf(mrow[j] - mi);
      mrow[j]  = mi;
      float p0 = __expf(s0[j] - mi);
      float p1 = __expf(s1[j] - mi);
      s0[j] = p0; s1[j] = p1;
      rs[j] = p0 + p1;
    }
#pragma unroll
    for (int off = 8; off >= 1; off >>= 1)
#pragma unroll
      for (int j = 0; j < 8; ++j) rs[j] += __shfl_xor(rs[j], off, 32);
#pragma unroll
    for (int j = 0; j < 8; ++j) lrow[j] = lrow[j] * alpha[j] + rs[j];
#pragma unroll
    for (int c = 0; c < 8; ++c)
#pragma unroll
      for (int j = 0; j < 8; ++j) o[c][j] *= alpha[j];

    // ---- P: C-layout -> LDS -> A-fragment (wave-private, dscnt fence) ----
#pragma unroll
    for (int j = 0; j < 8; ++j) {
      plds[(rbase + j) * 32 + coln]      = f2bf(s0[j]);
      plds[(rbase + j) * 32 + 16 + coln] = f2bf(s1[j]);
    }
    asm volatile("s_wait_dscnt 0" ::: "memory");
    bf16x16 pf = load_a_frag(plds, 32);

    // ---- O += P x V  (V^T tile rows contiguous over key positions) ----
#pragma unroll
    for (int c = 0; c < 8; ++c) {
      bf16x16 vf = load_b_frag16(vbuf + (size_t)(c * 16) * LDS_STRIDE, LDS_STRIDE);
      o[c] = wmma_bf16(pf, vf, o[c]);
    }
  }

  // ---- normalize + store context (bf16) ----
  __bf16* op = ctx + ((size_t)(b * S_LEN + qbase)) * QCOLS + h * HDIM;
#pragma unroll
  for (int j = 0; j < 8; ++j) lrow[j] = 1.0f / lrow[j];
#pragma unroll
  for (int c = 0; c < 8; ++c)
#pragma unroll
    for (int j = 0; j < 8; ++j)
      op[(size_t)(rbase + j) * QCOLS + c * 16 + coln] = f2bf(o[c][j] * lrow[j]);
}

// ---------------- workspace layout (bytes) ----------------------------------
constexpr size_t OFF_XBF = 0;                         // 4096*4096*2
constexpr size_t OFF_WQT = OFF_XBF + 33554432;        // 4096*4096*2
constexpr size_t OFF_WKT = OFF_WQT + 33554432;        // 1024*4096*2
constexpr size_t OFF_WVT = OFF_WKT + 8388608;         // 1024*4096*2
constexpr size_t OFF_WOT = OFF_WVT + 8388608;         // 4096*4096*2
constexpr size_t OFF_Q   = OFF_WOT + 33554432;        // 4096*4096*2
constexpr size_t OFF_K   = OFF_Q   + 33554432;        // 4096*1024*2
constexpr size_t OFF_VT  = OFF_K   + 8388608;         // 2*8*128*2048*2
constexpr size_t OFF_CTX = OFF_VT  + 8388608;         // 4096*4096*2
// total = 201326592 bytes (192 MB)

extern "C" void kernel_launch(void* const* d_in, const int* in_sizes, int n_in,
                              void* d_out, int out_size, void* d_ws, size_t ws_size,
                              hipStream_t stream) {
  const float* x    = (const float*)d_in[0];
  const float* cosb = (const float*)d_in[1];
  const float* sinb = (const float*)d_in[2];
  const float* wq   = (const float*)d_in[3];
  const float* wk   = (const float*)d_in[4];
  const float* wv   = (const float*)d_in[5];
  const float* wo   = (const float*)d_in[6];
  // d_in[7] = start_pos (always 0 here)

  char* ws = (char*)d_ws;
  __bf16* xbf = (__bf16*)(ws + OFF_XBF);
  __bf16* wqT = (__bf16*)(ws + OFF_WQT);
  __bf16* wkT = (__bf16*)(ws + OFF_WKT);
  __bf16* wvT = (__bf16*)(ws + OFF_WVT);
  __bf16* woT = (__bf16*)(ws + OFF_WOT);
  __bf16* Qb  = (__bf16*)(ws + OFF_Q);
  __bf16* Kb  = (__bf16*)(ws + OFF_K);
  __bf16* Vt  = (__bf16*)(ws + OFF_VT);
  __bf16* Ctx = (__bf16*)(ws + OFF_CTX);

  const int TPB = 256;
  int nx = BSROWS * DMODEL;            // 16777216
  k_f32_to_bf16<<<(nx + TPB - 1) / TPB, TPB, 0, stream>>>(x, xbf, nx);
  k_transpose_f32_bf16<<<(DMODEL * QCOLS + TPB - 1) / TPB, TPB, 0, stream>>>(wq, wqT, DMODEL, QCOLS);
  k_transpose_f32_bf16<<<(DMODEL * KVCOLS + TPB - 1) / TPB, TPB, 0, stream>>>(wk, wkT, DMODEL, KVCOLS);
  k_transpose_f32_bf16<<<(DMODEL * KVCOLS + TPB - 1) / TPB, TPB, 0, stream>>>(wv, wvT, DMODEL, KVCOLS);
  k_transpose_f32_bf16<<<(QCOLS * DMODEL + TPB - 1) / TPB, TPB, 0, stream>>>(wo, woT, QCOLS, DMODEL);

  // QKV projections (WMMA bf16, async-LDS double buffered)
  k_gemm_bf16<<<dim3(QCOLS / 128, BSROWS / 128), 256, 0, stream>>>(xbf, wqT, Qb, BSROWS, QCOLS, DMODEL, 0);
  k_gemm_bf16<<<dim3(KVCOLS / 128, BSROWS / 128), 256, 0, stream>>>(xbf, wkT, Kb, BSROWS, KVCOLS, DMODEL, 0);
  k_gemm_bf16<<<dim3(KVCOLS / 128, BSROWS / 128), 256, 0, stream>>>(xbf, wvT, Vt, BSROWS, KVCOLS, DMODEL, 1);

  // RoPE on Q and K
  int nq = BSROWS * NHEADS * 64;       // 8388608 pairs
  int nk = BSROWS * KVHEADS * 64;      // 2097152 pairs
  k_rope<<<(nq + TPB - 1) / TPB, TPB, 0, stream>>>(Qb, cosb, sinb, NHEADS, nq);
  k_rope<<<(nk + TPB - 1) / TPB, TPB, 0, stream>>>(Kb, cosb, sinb, KVHEADS, nk);

  // Causal flash attention: grid (q-tiles, heads, batch)
  k_flash_attn<<<dim3(S_LEN / 64, NHEADS, BATCH), 128, 0, stream>>>(Qb, Kb, Vt, Ctx);

  // Output projection -> f32 d_out
  k_gemm_bf16<<<dim3(DMODEL / 128, BSROWS / 128), 256, 0, stream>>>(Ctx, woT, (float*)d_out, BSROWS, DMODEL, QCOLS, 2);
}
